// DimeNetModel_48808008352217
// MI455X (gfx1250) — matrix operand order repaired and checked by reading, not compiled, not run on hardware
//
#include <hip/hip_runtime.h>

// ---------------------------------------------------------------------------
// DimeNet forward for MI455X (gfx1250).  All [*,128]x[128,128] contractions
// routed through v_wmma_f32_16x16x32_bf16 (wave32 WMMA, fp32 accum).
// 64-row blocks: each wave holds 4 accumulators so one B fragment feeds 4
// WMMAs from registers.  A/B staged into LDS with global_load_async_to_lds.
// Inner loop loads all 4 A-fragments into distinct registers before the
// 4-WMMA burst so ds waits don't serialize the matrix pipe.
// ---------------------------------------------------------------------------

typedef __bf16 bf16_t;
typedef __attribute__((ext_vector_type(16))) __bf16   bf16x16;
typedef __attribute__((ext_vector_type(8)))  float    v8f;
typedef __attribute__((ext_vector_type(4)))  unsigned u32x4;

union Frag16 { bf16x16 v; u32x4 q[2]; };

#define HDIM 128

// problem constants
static constexpr int NGRAPH = 2000;
static constexpr int APM    = 8;            // atoms per molecule
static constexpr int NNODE  = NGRAPH * APM;           // 16000
static constexpr int ELOC   = APM * (APM - 1);        // 56
static constexpr int NEDGE  = NGRAPH * ELOC;          // 112000
static constexpr int TLOC   = APM * (APM-1) * (APM-2);// 336
static constexpr int NTRI   = NGRAPH * TLOC;          // 672000

static constexpr float PI_F   = 3.14159265358979f;
static constexpr float ROOT2F = 1.41421356237310f;
static constexpr float Z1F    = 4.49340945790906f;
static constexpr float Z2F    = 7.72525183693771f;
static constexpr float N1F    = 6.5100946f;   // 1/sqrt(0.5*j2(Z1)^2)
static constexpr float N2F    = 11.0163066f;  // 1/sqrt(0.5*j2(Z2)^2)
static constexpr float Y00F   = 0.28209479177387814f;
static constexpr float Y10F   = 0.4886025119029199f;

__device__ __forceinline__ float silu_f(float x) {
    return x * (1.0f / (1.0f + __expf(-x)));
}

// ---------------------------------------------------------------------------
// Async global->LDS copy (CDNA5): per-lane 16B, tracked with ASYNCcnt.
// vdst VGPR carries the LDS byte address (low 32 bits of the shared-aperture
// flat address), vaddr the 64-bit global address.
// ---------------------------------------------------------------------------
__device__ __forceinline__ void async_g2l_b128(const void* gsrc, void* ldst) {
#if defined(__gfx1250__)
    unsigned long long ga = (unsigned long long)gsrc;
    unsigned           lo = (unsigned)(unsigned long long)ldst;
    asm volatile("global_load_async_to_lds_b128 %0, %1, off"
                 :: "v"(lo), "v"(ga) : "memory");
#else
    __builtin_memcpy(ldst, gsrc, 16);
#endif
}
__device__ __forceinline__ void wait_async_all() {
#if defined(__gfx1250__)
    asm volatile("s_wait_asynccnt 0" ::: "memory");
#endif
}

// ---------------------------------------------------------------------------
// Weight repack: fp32 [128,128] row-major -> bf16 WMMA-B fragment layout.
// For n-tile nt, k-step kk, lane L, element e:
//   k = kk*32 + (L>>4)*16 + e ,  n = nt*16 + (L&15)
// so a B fragment is 32 contiguous bytes per lane.
// ---------------------------------------------------------------------------
__global__ void k_pack_w(const float* __restrict__ src, bf16_t* __restrict__ dst,
                         int transpose) {
    int idx  = blockIdx.x * 256 + threadIdx.x;      // 0..16383
    int e    = idx & 15;
    int lane = (idx >> 4) & 31;
    int kk   = (idx >> 9) & 3;
    int nt   = idx >> 11;
    int k = kk * 32 + ((lane >> 4) << 4) + e;
    int n = (nt << 4) + (lane & 15);
    float v = transpose ? src[n * HDIM + k] : src[k * HDIM + n];
    dst[idx] = (bf16_t)v;
}

// ---------------------------------------------------------------------------
// Fused WMMA GEMM: Y[R,128] = epilogue( sum_c A_c[R,128] @ Wpk_c + ... )
// 256 threads = 8 waves; wave w owns n-tile w; 64-row strip per block
// (4 M-subtiles per wave).  LDS phase 1 = A(+B) staging, phase 2 = C staging.
// ---------------------------------------------------------------------------
constexpr int M_BIAS = 1, M_SILU = 2, M_MULRB = 4, M_ADDSRC = 8,
              M_F32ACC = 16, M_BF16OUT = 32;

template<int MODE, int KC>
__global__ __launch_bounds__(256) void k_gemm(
    const bf16_t* __restrict__ a0, const int* __restrict__ i0,
    const bf16_t* __restrict__ a1, const int* __restrict__ i1,
    const bf16_t* __restrict__ a2, const int* __restrict__ i2,
    const bf16_t* __restrict__ wp,                 // [KC][16384] packed B
    const float*  __restrict__ bias,               // [128]
    const float*  __restrict__ rbf2,               // [R,2]   (MULRB)
    const float*  __restrict__ rbw,                // [2,128] (MULRB)
    const bf16_t* __restrict__ addsrc,             // [R,128] (ADDSRC)
    bf16_t*       __restrict__ outb,               // [R,128] (BF16OUT)
    float*        __restrict__ accf)               // [R,128] (F32ACC)
{
    constexpr int  MT     = 4;
    constexpr int  AROWS  = 16 * MT;               // 64 rows per block
    constexpr int  APITCH = 136;                   // bf16/row (+8 pad: banks)
    constexpr bool BLDS   = (KC == 1);             // stage B in LDS too
    constexpr int  ABYTES = KC * AROWS * APITCH * 2;
    constexpr int  BBYTES = BLDS ? 16384 * 2 : 0;
    constexpr int  CBYTES = AROWS * HDIM * 4;
    constexpr int  SMEMB  = (ABYTES + BBYTES > CBYTES) ? (ABYTES + BBYTES) : CBYTES;
    __shared__ __align__(16) char smem[SMEMB];
    bf16_t* sA = (bf16_t*)smem;                    // phase 1
    float*  sC = (float*)smem;                     // phase 2 (aliased)

    const int tid  = threadIdx.x;
    const int lane = tid & 31;
    const int wave = tid >> 5;
    const int r0   = blockIdx.x * AROWS;

    // ---- phase 1: async-stage A strips (gathered rows) and B into LDS -----
    #pragma unroll
    for (int c = 0; c < KC; ++c) {
        const bf16_t* ab = (c == 0) ? a0 : (c == 1) ? a1 : a2;
        const int*    ai = (c == 0) ? i0 : (c == 1) ? i1 : i2;
        #pragma unroll
        for (int it = 0; it < (AROWS * 16) / 256; ++it) {   // 4 iters
            int task = it * 256 + tid;
            int row = task >> 4, seg = task & 15;
            int grow = ai ? ai[r0 + row] : (r0 + row);
            async_g2l_b128(ab + (size_t)grow * HDIM + seg * 8,
                           sA + c * AROWS * APITCH + row * APITCH + seg * 8);
        }
    }
    if constexpr (BLDS) {
        bf16_t* sB = (bf16_t*)(smem + ABYTES);
        #pragma unroll
        for (int it = 0; it < 8; ++it) {                    // 32KB
            int q = it * 256 + tid;                         // b128 index
            async_g2l_b128(wp + (size_t)q * 8, sB + (size_t)q * 8);
        }
    } else {
        // B streamed from global (L0-resident): warm it up
        __builtin_prefetch(wp + (size_t)tid * 64, 0, 1);
    }
    wait_async_all();
    __syncthreads();

    // ---- WMMA main loop: one B fragment feeds 4 M-subtile WMMAs ------------
    v8f zero8 = {0.f, 0.f, 0.f, 0.f, 0.f, 0.f, 0.f, 0.f};
    v8f acc[MT];
    #pragma unroll
    for (int m = 0; m < MT; ++m) acc[m] = zero8;

    const bf16_t* bbase;
    if constexpr (BLDS) bbase = (const bf16_t*)(smem + ABYTES);
    else                bbase = wp;

    const int arow  = lane & 15;
    const int khalf = lane >> 4;
    #pragma unroll
    for (int c = 0; c < KC; ++c) {
        #pragma unroll
        for (int kk = 0; kk < 4; ++kk) {
            // B fragment: 32 contiguous bytes per lane
            Frag16 bfr;
            const bf16_t* bp = bbase + (BLDS ? 0 : (size_t)c * 16384)
                             + ((size_t)((wave * 4 + kk) * 32 + lane)) * 16;
            bfr.q[0] = *(const u32x4*)(bp);
            bfr.q[1] = *(const u32x4*)(bp + 8);
            // all 4 A fragments into distinct registers (clustered ds_loads,
            // single wait) ...
            Frag16 af[MT];
            #pragma unroll
            for (int m = 0; m < MT; ++m) {
                const bf16_t* ap = sA + c * AROWS * APITCH
                                 + (m * 16 + arow) * APITCH + kk * 32 + khalf * 16;
                af[m].q[0] = *(const u32x4*)(ap);
                af[m].q[1] = *(const u32x4*)(ap + 8);
            }
            // ... then a back-to-back 4-WMMA burst
            #pragma unroll
            for (int m = 0; m < MT; ++m)
                acc[m] = __builtin_amdgcn_wmma_f32_16x16x32_bf16(
                    false, af[m].v, false, bfr.v, (short)0, acc[m], false, false);
        }
    }

    // ---- phase 2: stage accumulators, fused epilogue -----------------------
    __syncthreads();                       // done reading sA/sB; alias as sC
    {
        const int colb = (wave << 4) | (lane & 15);
        const int rb   = (lane >> 4) << 3;
        #pragma unroll
        for (int m = 0; m < MT; ++m)
            #pragma unroll
            for (int i = 0; i < 8; ++i)
                sC[(m * 16 + rb + i) * HDIM + colb] = acc[m][i];
    }
    __syncthreads();
    #pragma unroll
    for (int it = 0; it < (AROWS * 16) / 256; ++it) {
        const int task = it * 256 + tid;
        const int row  = task >> 4;
        const int c0   = (task & 15) << 3;
        const int grow = r0 + row;
        float rv0 = 0.f, rv1 = 0.f;
        if (MODE & M_MULRB) { rv0 = rbf2[grow * 2]; rv1 = rbf2[grow * 2 + 1]; }
        #pragma unroll
        for (int j = 0; j < 8; ++j) {
            int col = c0 + j;
            float y = sC[row * HDIM + col];
            if (MODE & M_BIAS)   y += bias[col];
            if (MODE & M_SILU)   y  = silu_f(y);
            if (MODE & M_MULRB)  y *= rv0 * rbw[col] + rv1 * rbw[HDIM + col];
            if (MODE & M_ADDSRC) y += (float)addsrc[(size_t)grow * HDIM + col];
            if (MODE & M_F32ACC) accf[(size_t)grow * HDIM + col] += y;
            if (MODE & M_BF16OUT) outb[(size_t)grow * HDIM + col] = (bf16_t)y;
        }
    }
}

// ---------------------------------------------------------------------------
// Elementwise / gather kernels
// ---------------------------------------------------------------------------
__global__ void k_geom(const float* __restrict__ pos, const int* __restrict__ ei,
                       const int* __restrict__ ej, float* __restrict__ rbf,
                       float* __restrict__ sb, int E) {
    int e = blockIdx.x * 256 + threadIdx.x;
    if (e >= E) return;
    int i = ei[e], j = ej[e];
    float dx = pos[i*3+0] - pos[j*3+0];
    float dy = pos[i*3+1] - pos[j*3+1];
    float dz = pos[i*3+2] - pos[j*3+2];
    float dist = sqrtf(dx*dx + dy*dy + dz*dz);
    float d  = dist * 0.2f;                       // /CUTOFF
    float d2 = d*d, x5 = d2*d2*d;
    float env = (d < 1.0f)
        ? (1.0f/d - 28.0f*x5 + 48.0f*x5*d - 21.0f*x5*d2) : 0.0f;
    rbf[e*2+0] = env * sinf(PI_F * d);
    rbf[e*2+1] = env * sinf(2.0f * PI_F * d);
    float u1 = Z1F * d, u2 = Z2F * d;
    float j1a = sinf(u1)/(u1*u1) - cosf(u1)/u1;
    float j1b = sinf(u2)/(u2*u2) - cosf(u2)/u2;
    sb[e*4+0] = env * ROOT2F * sinf(PI_F * d) / d;
    sb[e*4+1] = env * ROOT2F * sinf(2.0f * PI_F * d) / d;
    sb[e*4+2] = env * N1F * j1a;
    sb[e*4+3] = env * N2F * j1b;
}

// per-triplet angular basis, pre-contracted with each block's lin_sbf [4,1]
__global__ void k_tri(const float* __restrict__ pos,
                      const int* __restrict__ ti, const int* __restrict__ tj,
                      const int* __restrict__ tk, const int* __restrict__ tkj,
                      const float* __restrict__ sb,
                      const float* __restrict__ s0, const float* __restrict__ s1,
                      const float* __restrict__ s2,
                      float* __restrict__ sf, int T) {
    int t = blockIdx.x * 256 + threadIdx.x;
    if (t >= T) return;
    int i = ti[t], j = tj[t], k = tk[t];
    float jx = pos[j*3+0]-pos[i*3+0], jy = pos[j*3+1]-pos[i*3+1], jz = pos[j*3+2]-pos[i*3+2];
    float kx = pos[k*3+0]-pos[i*3+0], ky = pos[k*3+1]-pos[i*3+1], kz = pos[k*3+2]-pos[i*3+2];
    float aa = jx*kx + jy*ky + jz*kz;
    float cx = jy*kz - jz*ky, cy = jz*kx - jx*kz, cz = jx*ky - jy*kx;
    float bb = sqrtf(cx*cx + cy*cy + cz*cz);
    float ang = atan2f(bb, aa);
    float c0 = Y00F;
    float c1 = Y10F * cosf(ang);
    int ekj = tkj[t];
    float f0 = sb[ekj*4+0]*c0, f1 = sb[ekj*4+1]*c0;
    float f2 = sb[ekj*4+2]*c1, f3 = sb[ekj*4+3]*c1;
    sf[t]       = f0*s0[0] + f1*s0[1] + f2*s0[2] + f3*s0[3];
    sf[T + t]   = f0*s1[0] + f1*s1[1] + f2*s1[2] + f3*s1[3];
    sf[2*T + t] = f0*s2[0] + f1*s2[1] + f2*s2[2] + f3*s2[3];
}

__global__ void k_hnodes(const int* __restrict__ z, const float* __restrict__ emb,
                         bf16_t* __restrict__ hb, int n) {
    int idx = blockIdx.x * 256 + threadIdx.x;
    if (idx >= n) return;
    int nn = idx >> 7, h = idx & 127;
    hb[idx] = (bf16_t)emb[z[nn] * HDIM + h];
}

__global__ void k_re(const float* __restrict__ rbf, const float* __restrict__ w,
                     const float* __restrict__ b, bf16_t* __restrict__ re, int n) {
    int idx = blockIdx.x * 256 + threadIdx.x;
    if (idx >= n) return;
    int e = idx >> 7, h = idx & 127;
    float y = rbf[e*2] * w[h] + rbf[e*2+1] * w[HDIM + h] + b[h];
    re[idx] = (bf16_t)silu_f(y);
}

// triplet aggregation: 6 consecutive triplets share idx_ji -> one block each
__global__ void k_agg(const bf16_t* __restrict__ xkj, const float* __restrict__ sf,
                      const int* __restrict__ tkj, const int* __restrict__ tji,
                      bf16_t* __restrict__ agg) {
    int t0 = blockIdx.x * 6;
    int h  = threadIdx.x;
    int e  = tji[t0];
    float s = 0.f;
    #pragma unroll
    for (int q = 0; q < 6; ++q) {
        float w = sf[t0 + q];
        int ek  = tkj[t0 + q];
        s += w * (float)xkj[(size_t)ek * HDIM + h];
    }
    agg[(size_t)e * HDIM + h] = (bf16_t)s;
}

// out-block head: nodes[n] = sum over 7 incoming edges of (rbf@lin_rbf)*x
// complete-graph edge id: e_loc(src,dst) = src*7 + dst - (dst>src)
__global__ void k_outgather(const bf16_t* __restrict__ x, const float* __restrict__ rbf2,
                            const float* __restrict__ w, bf16_t* __restrict__ nodes) {
    int n = blockIdx.x, h = threadIdx.x;
    int g = n >> 3, i = n & 7;
    float w0 = w[h], w1 = w[HDIM + h];
    float s = 0.f;
    #pragma unroll
    for (int j = 0; j < 8; ++j) {
        if (j == i) continue;
        int eloc = j * 7 + i - (i > j ? 1 : 0);
        size_t e = (size_t)g * ELOC + eloc;
        float rb = rbf2[e*2] * w0 + rbf2[e*2+1] * w1;
        s += rb * (float)x[e * HDIM + h];
    }
    nodes[(size_t)n * HDIM + h] = (bf16_t)s;
}

__global__ void k_zero(float* __restrict__ p, int n) {
    int idx = blockIdx.x * 256 + threadIdx.x;
    if (idx < n) p[idx] = 0.f;
}

// graph readout + head: out[g] = sum_{a,h} P[g*8+a,h]*head_w[h] + head_b
__global__ void k_head(const float* __restrict__ P, const float* __restrict__ hw,
                       const float* __restrict__ hb, float* __restrict__ out) {
    __shared__ float red[HDIM];
    int g = blockIdx.x, h = threadIdx.x;
    float s = 0.f;
    #pragma unroll
    for (int a = 0; a < APM; ++a) s += P[((size_t)g * APM + a) * HDIM + h];
    red[h] = s * hw[h];
    __syncthreads();
    for (int st = 64; st > 0; st >>= 1) {
        if (h < st) red[h] += red[h + st];
        __syncthreads();
    }
    if (h == 0) out[g] = red[0] + hb[0];
}

// ---------------------------------------------------------------------------
extern "C" void kernel_launch(void* const* d_in, const int* in_sizes, int n_in,
                              void* d_out, int out_size, void* d_ws, size_t ws_size,
                              hipStream_t stream) {
    (void)in_sizes; (void)n_in; (void)out_size; (void)ws_size;

    const int*   z      = (const int*)  d_in[0];
    const float* pos    = (const float*)d_in[1];
    const int*   edge_i = (const int*)  d_in[3];
    const int*   edge_j = (const int*)  d_in[4];
    const int*   idx_i  = (const int*)  d_in[5];
    const int*   idx_j  = (const int*)  d_in[6];
    const int*   idx_k  = (const int*)  d_in[7];
    const int*   idx_kj = (const int*)  d_in[8];
    const int*   idx_ji = (const int*)  d_in[9];
    auto F = [&](int i) { return (const float*)d_in[i]; };

    // params flattened in dict insertion order
    const int P_EMB = 11, P_ERBW = 12, P_ERBB = 13, P_ELINW = 14, P_ELINB = 15;
    auto PO  = [](int ob) { return 16 + 4 * ob; };   // lin_rbf, lins_w, lins_b, lin_out
    auto PIB = [](int b)  { return 32 + 21 * b; };   // int block base
    const int P_HEADW = 95, P_HEADB = 96;

    // workspace carve-up
    char* ws = (char*)d_ws;
    size_t off = 0;
    auto carve = [&](size_t bytes) {
        char* p = ws + off;
        off = (off + bytes + 255) & ~(size_t)255;
        return p;
    };
    bf16_t* wpk  = (bf16_t*)carve((size_t)49 * 16384 * sizeof(bf16_t));
    float*  rbf  = (float*) carve((size_t)NEDGE * 2 * 4);
    float*  sb   = (float*) carve((size_t)NEDGE * 4 * 4);
    float*  sf   = (float*) carve((size_t)3 * NTRI * 4);
    float*  Pacc = (float*) carve((size_t)NNODE * HDIM * 4);
    bf16_t* xb   = (bf16_t*)carve((size_t)NEDGE * HDIM * 2);
    bf16_t* xji  = (bf16_t*)carve((size_t)NEDGE * HDIM * 2);  // also hh
    bf16_t* xkj  = (bf16_t*)carve((size_t)NEDGE * HDIM * 2);  // also r_e
    bf16_t* tb   = (bf16_t*)carve((size_t)NEDGE * HDIM * 2);  // also agg
    bf16_t* nb0  = (bf16_t*)carve((size_t)NNODE * HDIM * 2);  // also h-nodes
    bf16_t* nb1  = (bf16_t*)carve((size_t)NNODE * HDIM * 2);

    // ---- pack all weight matrices into WMMA-B layout -----------------------
    int slot = 0;
    auto pack = [&](const float* src, int tr) {
        bf16_t* d = wpk + (size_t)(slot++) * 16384;
        k_pack_w<<<dim3(64), dim3(256), 0, stream>>>(src, d, tr);
        return d;
    };
    bf16_t* wp_emb = pack(F(P_ELINW), 0);            // 3 consecutive K-chunks
    pack(F(P_ELINW) + 16384, 0);
    pack(F(P_ELINW) + 32768, 0);
    bf16_t *wp_ji[3], *wp_kj[3], *wp_msg[3], *wp_bw1[3], *wp_bw2[3], *wp_lin[3];
    bf16_t *wp_aw1[3][2], *wp_aw2[3][2];
    for (int b = 0; b < 3; ++b) {
        int pb = PIB(b);
        wp_kj[b]  = pack(F(pb + 2), 0);
        wp_ji[b]  = pack(F(pb + 4), 0);
        wp_msg[b] = pack(F(pb + 6), 1);              // W[:,0,:]^T
        wp_bw1[b] = pack(F(pb + 7), 0);
        wp_bw2[b] = pack(F(pb + 9), 0);
        wp_lin[b] = pack(F(pb + 11), 0);
        wp_aw1[b][0] = pack(F(pb + 13), 0);
        wp_aw2[b][0] = pack(F(pb + 15), 0);
        wp_aw1[b][1] = pack(F(pb + 17), 0);
        wp_aw2[b][1] = pack(F(pb + 19), 0);
    }
    bf16_t *wp_ol[4][3], *wp_oo[4];
    for (int ob = 0; ob < 4; ++ob) {
        int po = PO(ob);
        for (int j = 0; j < 3; ++j) wp_ol[ob][j] = pack(F(po + 1) + (size_t)j * 16384, 0);
        wp_oo[ob] = pack(F(po + 3), 0);
    }

    const int ES = NEDGE / 64, NS = NNODE / 64;      // 1750 / 250 row blocks
    const bf16_t* NB = nullptr; const int* NI = nullptr;
    const float* NF = nullptr;  bf16_t* NO = nullptr; float* NA = nullptr;

    // ---- geometry / basis --------------------------------------------------
    k_geom<<<dim3((NEDGE + 255) / 256), dim3(256), 0, stream>>>(pos, edge_i, edge_j, rbf, sb, NEDGE);
    k_tri <<<dim3((NTRI + 255) / 256),  dim3(256), 0, stream>>>(
        pos, idx_i, idx_j, idx_k, idx_kj, sb,
        F(PIB(0) + 1), F(PIB(1) + 1), F(PIB(2) + 1), sf, NTRI);
    k_hnodes<<<dim3((NNODE * HDIM + 255) / 256), dim3(256), 0, stream>>>(z, F(P_EMB), nb0, NNODE * HDIM);
    k_re   <<<dim3((NEDGE * HDIM + 255) / 256), dim3(256), 0, stream>>>(rbf, F(P_ERBW), F(P_ERBB), xkj, NEDGE * HDIM);

    // ---- embedding block: x = silu([h_i | h_j | r_e] @ Wemb + b) -----------
    k_gemm<M_BIAS | M_SILU | M_BF16OUT, 3><<<dim3(ES), dim3(256), 0, stream>>>(
        nb0, edge_i, nb0, edge_j, xkj, NI, wp_emb, F(P_ELINB), NF, NF, NB, xb, NA);

    k_zero<<<dim3((NNODE * HDIM + 255) / 256), dim3(256), 0, stream>>>(Pacc, NNODE * HDIM);

    auto out_block = [&](int ob) {
        int po = PO(ob);
        k_outgather<<<dim3(NNODE), dim3(HDIM), 0, stream>>>(xb, rbf, F(po + 0), nb0);
        bf16_t* src = nb0; bf16_t* dst = nb1;
        for (int j = 0; j < 3; ++j) {
            k_gemm<M_BIAS | M_SILU | M_BF16OUT, 1><<<dim3(NS), dim3(256), 0, stream>>>(
                src, NI, NB, NI, NB, NI, wp_ol[ob][j], F(po + 2) + j * HDIM,
                NF, NF, NB, dst, NA);
            bf16_t* t = src; src = dst; dst = t;
        }
        k_gemm<M_F32ACC, 1><<<dim3(NS), dim3(256), 0, stream>>>(
            src, NI, NB, NI, NB, NI, wp_oo[ob], NF, NF, NF, NB, NO, Pacc);
    };
    out_block(0);

    // ---- interaction blocks ------------------------------------------------
    for (int b = 0; b < 3; ++b) {
        int pb = PIB(b);
        // x_ji = silu(x @ Wji + b)
        k_gemm<M_BIAS | M_SILU | M_BF16OUT, 1><<<dim3(ES), dim3(256), 0, stream>>>(
            xb, NI, NB, NI, NB, NI, wp_ji[b], F(pb + 5), NF, NF, NB, xji, NA);
        // x_kj = silu(x @ Wkj + b) * (rbf @ lin_rbf)
        k_gemm<M_BIAS | M_SILU | M_MULRB | M_BF16OUT, 1><<<dim3(ES), dim3(256), 0, stream>>>(
            xb, NI, NB, NI, NB, NI, wp_kj[b], F(pb + 3), rbf, F(pb + 0), NB, xkj, NA);
        // agg[e] = sum over its 6 triplets of sf[t] * x_kj[idx_kj[t]]
        k_agg<<<dim3(NTRI / 6), dim3(HDIM), 0, stream>>>(xkj, sf + (size_t)b * NTRI, idx_kj, idx_ji, tb);
        // hh = x_ji + agg @ W_bilinear^T   (in-place add into xji)
        k_gemm<M_ADDSRC | M_BF16OUT, 1><<<dim3(ES), dim3(256), 0, stream>>>(
            tb, NI, NB, NI, NB, NI, wp_msg[b], NF, NF, NF, xji, xji, NA);
        // before-residual
        k_gemm<M_BIAS | M_SILU | M_BF16OUT, 1><<<dim3(ES), dim3(256), 0, stream>>>(
            xji, NI, NB, NI, NB, NI, wp_bw1[b], F(pb + 8), NF, NF, NB, tb, NA);
        k_gemm<M_BIAS | M_SILU | M_ADDSRC | M_BF16OUT, 1><<<dim3(ES), dim3(256), 0, stream>>>(
            tb, NI, NB, NI, NB, NI, wp_bw2[b], F(pb + 10), NF, NF, xji, xji, NA);
        // x_new = silu(hh @ Wlin + b) + x   (writes xb; A is xji so safe)
        k_gemm<M_BIAS | M_SILU | M_ADDSRC | M_BF16OUT, 1><<<dim3(ES), dim3(256), 0, stream>>>(
            xji, NI, NB, NI, NB, NI, wp_lin[b], F(pb + 12), NF, NF, xb, xb, NA);
        // two after-residuals (in-place on xb)
        for (int r = 0; r < 2; ++r) {
            const float* b1 = F(pb + 14 + 4 * r);
            const float* b2 = F(pb + 16 + 4 * r);
            k_gemm<M_BIAS | M_SILU | M_BF16OUT, 1><<<dim3(ES), dim3(256), 0, stream>>>(
                xb, NI, NB, NI, NB, NI, wp_aw1[b][r], b1, NF, NF, NB, tb, NA);
            k_gemm<M_BIAS | M_SILU | M_ADDSRC | M_BF16OUT, 1><<<dim3(ES), dim3(256), 0, stream>>>(
                tb, NI, NB, NI, NB, NI, wp_aw2[b][r], b2, NF, NF, xb, xb, NA);
        }
        out_block(b + 1);
    }

    // ---- readout -----------------------------------------------------------
    k_head<<<dim3(NGRAPH), dim3(HDIM), 0, stream>>>(Pacc, F(P_HEADW), F(P_HEADB), (float*)d_out);
}
